// CustomConv_1417339208184
// MI455X (gfx1250) — compile-verified, matrix operand order checked by reading, not run on hardware
//
#include <hip/hip_runtime.h>

// ---------------------------------------------------------------------------
// Sizes from the reference
// ---------------------------------------------------------------------------
#define A_AP   128
#define U_UE   8
#define ANT    64
#define D2     64
#define H      256
#define NUE    (A_AP * U_UE)        // 1024
#define E_INT  (A_AP * NUE)         // 131072
#define K2IN   (4 * ANT)            // 256  (input dim of mlp2)
#define K1RAW  (1 + 2 * ANT + D2)   // 193  (input dim of mlp1)
#define K1PAD  224                  // padded to 7 k-tiles of 32
// Strides (in halves): multiples of 8, chosen to break the 512B-row-stride
// LDS bank pattern for the per-lane b128 fragment loads.
#define XS2    272                  // x-tile / weight K-stride for K=256
#define XS1    240                  // stride for K1PAD=224
#define HSS    40                   // per-wave hidden scratch stride (K=32)

// k2 LDS layout (in halves), all constant offsets from one addrspace(3) root:
#define K2_XT1   (16 * XS2)                 // second x-tile buffer
#define K2_HS    (2 * 16 * XS2)             // hidden scratch base
#define K2_OUT   (K2_HS + 8 * 16 * HSS)     // outred (float) base, in halves
// k4 LDS layout (in halves):
#define K4_HS    (16 * XS1)
#define K4_OUT   (K4_HS + 8 * 16 * HSS)

typedef _Float16 h16 __attribute__((ext_vector_type(16)));
typedef _Float16 h8  __attribute__((ext_vector_type(8)));
typedef _Float16 h4  __attribute__((ext_vector_type(4)));
typedef float    f8  __attribute__((ext_vector_type(8)));

#define WMMA_F16(a, b, c) \
  __builtin_amdgcn_wmma_f32_16x16x32_f16(false, (a), false, (b), (short)0, (c), false, false)

// Per-lane fragment address for the CDNA5 16-bit 16x16x32 A/B layouts:
//   lanes 0-15 : rows 0-15, K = kbase+0..7 (V0-3) and kbase+16..23 (V4-7)
//   lanes 16-31: rows 0-15, K = kbase+8..15 (V0-3) and kbase+24..31 (V4-7)
// (B fragments use the same pattern when the matrix is stored [N][K].)
__device__ __forceinline__ const _Float16*
frag_ptr(const _Float16* base, int row0, int stride, int kbase, int lane) {
  int r  = lane & 15;
  int kh = (lane >> 4) & 1;
  return base + (row0 + r) * stride + kbase + kh * 8;
}

// Works for both LDS and global pointers: emits ds_load_b128 / global_load_b128.
__device__ __forceinline__ h16 frag_load(const _Float16* p) {
  h8 lo = *(const h8*)(p);
  h8 hi = *(const h8*)(p + 16);
  return __builtin_shufflevector(lo, hi, 0, 1, 2, 3, 4, 5, 6, 7,
                                 8, 9, 10, 11, 12, 13, 14, 15);
}

// ---------------------------------------------------------------------------
// Kernel 1a: per-AP sum of power vectors (f32 sum -> f16), ssh[a][2*ANT]
// ---------------------------------------------------------------------------
__global__ __launch_bounds__(64) void k1_ss(const float* __restrict__ pv_re,
                                            const float* __restrict__ pv_im,
                                            _Float16* __restrict__ ssh) {
  int a = blockIdx.x, t = threadIdx.x;  // 128 blocks x 64 threads
  float sr = 0.f, si = 0.f;
  for (int u = 0; u < U_UE; ++u) {
    sr += pv_re[(a * U_UE + u) * ANT + t];
    si += pv_im[(a * U_UE + u) * ANT + t];
  }
  ssh[a * 128 + t]      = (_Float16)sr;
  ssh[a * 128 + 64 + t] = (_Float16)si;
}

// ---------------------------------------------------------------------------
// Kernel 1b: transpose+convert all weights to f16 [N][K] layout; zero mlp_ue
// ---------------------------------------------------------------------------
__global__ __launch_bounds__(256) void k1_wt(const float* __restrict__ W2a,
                                             const float* __restrict__ W2b,
                                             const float* __restrict__ W1a,
                                             const float* __restrict__ W1b,
                                             _Float16* __restrict__ w2aT,
                                             _Float16* __restrict__ w2bT,
                                             _Float16* __restrict__ w1aT,
                                             _Float16* __restrict__ w1bT,
                                             float* __restrict__ mlp_ue) {
  const int N0 = H * K2IN;        // 65536  W2a: [K=256][N=256] -> [n][k]
  const int N1 = D2 * H;          // 16384  W2b: [K=256][N=64]  -> [n][k]
  const int N2 = H * K1PAD;       // 57344  W1a: [K=193][N=256] -> [n][k<=224]
  const int N3 = (2 * ANT) * H;   // 32768  W1b: [K=256][N=128] -> [n][k]
  const int N4 = NUE * D2;        // 65536  zero accumulation buffer
  int i = blockIdx.x * 256 + threadIdx.x;
  if (i < N0) {
    int n = i >> 8, k = i & 255;
    w2aT[n * XS2 + k] = (_Float16)W2a[k * H + n];
  } else if (i < N0 + N1) {
    int j = i - N0, n = j >> 8, k = j & 255;
    w2bT[n * XS2 + k] = (_Float16)W2b[k * D2 + n];
  } else if (i < N0 + N1 + N2) {
    int j = i - N0 - N1, n = j / K1PAD, k = j % K1PAD;
    w1aT[n * XS1 + k] = (k < K1RAW) ? (_Float16)W1a[k * H + n] : (_Float16)0.f;
  } else if (i < N0 + N1 + N2 + N3) {
    int j = i - N0 - N1 - N2, n = j >> 8, k = j & 255;
    w1bT[n * XS1 + k] = (_Float16)W1b[k * (2 * ANT) + n];
  } else if (i < N0 + N1 + N2 + N3 + N4) {
    mlp_ue[i - N0 - N1 - N2 - N3] = 0.f;
  }
}

// ---------------------------------------------------------------------------
// Kernel 2: big edge MLP + segment-sum (the 21.5 GFLOP hot spot).
// grid = 256 blocks: (u-block of 16 UEs) x (AP quarter of 32 APs), 8 waves.
// Wave w owns hidden cols [32w, 32w+32):
//   - W2a/W2b slices hoisted into registers (20 frags = 160 VGPRs), from L2.
//   - Double-buffered x-tile selected by an *integer LDS offset* (not a
//     pointer select) so all tile accesses stay addrspace(3) -> ds_load_b128.
//   - Software pipeline: next tile staged in registers during the WMMAs.
//   - GEMM2 is K-split: each wave consumes its own hidden slice and
//     accumulates its partial [16x64] output over all 32 APs in registers.
// ---------------------------------------------------------------------------
__global__ __launch_bounds__(256) void k2_mlp2(const float* __restrict__ pl_re,
                                               const float* __restrict__ pl_im,
                                               const _Float16* __restrict__ ssh,
                                               const _Float16* __restrict__ w2aT,
                                               const _Float16* __restrict__ w2bT,
                                               const float* __restrict__ b2a,
                                               float* __restrict__ mlp_ue) {
  extern __shared__ _Float16 lds2[];   // [2*16*XS2 | 8*16*HSS | 16*64 f32]
  float* outred = (float*)(lds2 + K2_OUT);

  const int tid = threadIdx.x, lane = tid & 31, wv = tid >> 5;
  const int ub = blockIdx.x & 63, as = blockIdx.x >> 6;
  const int u0 = ub * 16, a0 = as * 32;

  // ---- hoist loop-invariant B fragments: global(L2) -> VGPRs ----
  h16 bf1a[8], bf1b[8], bf2[4];
#pragma unroll
  for (int kt = 0; kt < 8; ++kt) {
    bf1a[kt] = frag_load(frag_ptr(w2aT, wv * 32,      XS2, kt * 32, lane));
    bf1b[kt] = frag_load(frag_ptr(w2aT, wv * 32 + 16, XS2, kt * 32, lane));
  }
#pragma unroll
  for (int nt = 0; nt < 4; ++nt)
    bf2[nt] = frag_load(frag_ptr(w2bT, nt * 16, XS2, wv * 32, lane));

  for (int i = tid; i < 16 * 64; i += 256) outred[i] = 0.f;

  const float bias0 = b2a[wv * 32 + (lane & 15)];
  const float bias1 = b2a[wv * 32 + 16 + (lane & 15)];

  const f8 zero8 = {0.f, 0.f, 0.f, 0.f, 0.f, 0.f, 0.f, 0.f};
  f8 acc2[4];
#pragma unroll
  for (int nt = 0; nt < 4; ++nt) acc2[nt] = zero8;

  // x-tile build indices (fixed per thread)
  const int row = (tid * 4) >> 6, col = (tid * 4) & 63;
  const int t = tid & 63, rr = tid >> 6;

  // ---- stage tile 0 into buffer 0 ----
  {
    const size_t gb4 = (size_t)(a0 * NUE + u0) * ANT / 4;
    float4 vr = ((const float4*)pl_re)[gb4 + tid];
    float4 vi = ((const float4*)pl_im)[gb4 + tid];
    _Float16 sr = ssh[a0 * 128 + t], si = ssh[a0 * 128 + 64 + t];
    h4 hr = {(_Float16)vr.x, (_Float16)vr.y, (_Float16)vr.z, (_Float16)vr.w};
    h4 hi = {(_Float16)vi.x, (_Float16)vi.y, (_Float16)vi.z, (_Float16)vi.w};
    *(h4*)(lds2 + row * XS2 + col)       = hr;
    *(h4*)(lds2 + row * XS2 + 128 + col) = hi;
    for (int r = rr; r < 16; r += 4) {
      lds2[r * XS2 + 64 + t]  = sr;
      lds2[r * XS2 + 192 + t] = si;
    }
  }
  __syncthreads();

  for (int it = 0; it < 32; ++it) {
    const int coff = (it & 1) * K2_XT1;        // current buffer (LDS offset)
    const int noff = coff ^ K2_XT1;            // next buffer

    // ---- stage next tile into registers (overlaps with WMMA below) ----
    const bool hn = (it + 1 < 32);
    float4 vr = {0.f, 0.f, 0.f, 0.f}, vi = {0.f, 0.f, 0.f, 0.f};
    _Float16 sr = (_Float16)0.f, si = (_Float16)0.f;
    if (hn) {
      const int an = a0 + it + 1;
      const size_t gb4 = (size_t)(an * NUE + u0) * ANT / 4;
      vr = ((const float4*)pl_re)[gb4 + tid];
      vi = ((const float4*)pl_im)[gb4 + tid];
      sr = ssh[an * 128 + t];
      si = ssh[an * 128 + 64 + t];
    }
    if (it + 2 < 32) {  // global_prefetch_b8, no counter cost
      const size_t gb = (size_t)((a0 + it + 2) * NUE + u0) * ANT + tid * 4;
      __builtin_prefetch(pl_re + gb, 0, 3);
      __builtin_prefetch(pl_im + gb, 0, 3);
    }

    // ---- GEMM1: 2 hidden n-tiles per wave, K = 8 k-tiles ----
    f8 acc0 = zero8, acc1 = zero8;
#pragma unroll
    for (int kt = 0; kt < 8; ++kt) {
      h16 aF = frag_load(frag_ptr(lds2 + coff, 0, XS2, kt * 32, lane));
      acc0 = WMMA_F16(aF, bf1a[kt], acc0);
      acc1 = WMMA_F16(aF, bf1b[kt], acc1);
    }

    // ---- bias + relu -> per-wave hidden scratch (f16) ----
    const int hoff = K2_HS + wv * 16 * HSS;
    const int r0 = (lane < 16) ? 0 : 8, nl = lane & 15;
#pragma unroll
    for (int i = 0; i < 8; ++i) {
      float v0 = acc0[i] + bias0; v0 = v0 > 0.f ? v0 : 0.f;
      float v1 = acc1[i] + bias1; v1 = v1 > 0.f ? v1 : 0.f;
      lds2[hoff + (r0 + i) * HSS + nl]      = (_Float16)v0;
      lds2[hoff + (r0 + i) * HSS + 16 + nl] = (_Float16)v1;
    }

    // ---- GEMM2 (K-split), accumulated across the AP loop in registers ----
    h16 a2 = frag_load(frag_ptr(lds2 + hoff, 0, HSS, 0, lane));
#pragma unroll
    for (int nt = 0; nt < 4; ++nt)
      acc2[nt] = WMMA_F16(a2, bf2[nt], acc2[nt]);

    // ---- commit staged tile to the alternate buffer ----
    if (hn) {
      h4 hr = {(_Float16)vr.x, (_Float16)vr.y, (_Float16)vr.z, (_Float16)vr.w};
      h4 hi = {(_Float16)vi.x, (_Float16)vi.y, (_Float16)vi.z, (_Float16)vi.w};
      *(h4*)(lds2 + noff + row * XS2 + col)       = hr;
      *(h4*)(lds2 + noff + row * XS2 + 128 + col) = hi;
      for (int r = rr; r < 16; r += 4) {
        lds2[noff + r * XS2 + 64 + t]  = sr;
        lds2[noff + r * XS2 + 192 + t] = si;
      }
    }
    __syncthreads();
  }

  // ---- cross-wave K reduction in LDS, then global segment accumulate ----
  {
    const int r0 = (lane < 16) ? 0 : 8, nl = lane & 15;
#pragma unroll
    for (int nt = 0; nt < 4; ++nt)
#pragma unroll
      for (int i = 0; i < 8; ++i)
        atomicAdd(&outred[(r0 + i) * 64 + nt * 16 + nl], acc2[nt][i]);
  }
  __syncthreads();
  for (int i = tid; i < 16 * 64; i += 256) {
    int m = i >> 6, d = i & 63;
    atomicAdd(&mlp_ue[(u0 + m) * D2 + d], outred[i]);
  }
}

// ---------------------------------------------------------------------------
// Kernel 3: in_infer[e] = sum_{u' != e%8} |<conj(pv[a][u']), pl_dl[e]>|^2
// ---------------------------------------------------------------------------
__global__ __launch_bounds__(256) void k3_ininf(const float* __restrict__ pv_re,
                                                const float* __restrict__ pv_im,
                                                const float* __restrict__ pld_re,
                                                const float* __restrict__ pld_im,
                                                float* __restrict__ in_inf) {
  __shared__ float red[256];
  int tid = threadIdx.x;
  int gid = blockIdx.x * 256 + tid;   // 32 blocks
  int e = gid >> 3, up = gid & 7, a = e >> 3;
  const float* pr = pv_re + (a * U_UE + up) * ANT;
  const float* pi = pv_im + (a * U_UE + up) * ANT;
  const float* qr = pld_re + e * ANT;
  const float* qi = pld_im + e * ANT;
  float ar = 0.f, ai = 0.f;
  for (int t = 0; t < ANT; ++t) {
    float p0 = pr[t], p1 = pi[t], q0 = qr[t], q1 = qi[t];
    ar += p0 * q0 + p1 * q1;   // Re(conj(p) * q)
    ai += p0 * q1 - p1 * q0;   // Im(conj(p) * q)
  }
  float nrm = ar * ar + ai * ai;
  if (up == (e & 7)) nrm = 0.f;
  red[tid] = nrm;
  __syncthreads();
  if ((tid & 7) == 0) {
    float s = 0.f;
    for (int j = 0; j < 8; ++j) s += red[tid + j];
    in_inf[e] = s;
  }
}

// ---------------------------------------------------------------------------
// Kernel 4: per-UE MLP (1024 rows). 64 blocks x 1 row-tile x 8 waves.
// Weight fragments are each used exactly once per block, so they are loaded
// straight from global (L2) at the point of use — no LDS staging.
// ---------------------------------------------------------------------------
__global__ __launch_bounds__(256) void k4_mlp1(const float* __restrict__ pld_re,
                                               const float* __restrict__ pld_im,
                                               const float* __restrict__ in_inf,
                                               const float* __restrict__ mlp_ue,
                                               const float* __restrict__ b2b,
                                               const float* __restrict__ b1a,
                                               const float* __restrict__ b1b,
                                               const _Float16* __restrict__ w1aT,
                                               const _Float16* __restrict__ w1bT,
                                               float* __restrict__ ue_re,
                                               float* __restrict__ ue_im) {
  extern __shared__ _Float16 lds4[];   // [16*XS1 | 8*16*HSS | 16*128 f32]
  float* outred = (float*)(lds4 + K4_OUT);

  const int tid = threadIdx.x, lane = tid & 31, wv = tid >> 5;
  const int e0 = blockIdx.x * 16;

  for (int i = tid; i < 16 * 128; i += 256) outred[i] = 0.f;

  // feat row: [in_infer | pl_dl.re(64) | pl_dl.im(64) | mlp_ue + 128*b2b | 0-pad]
  for (int idx = tid; idx < 16 * XS1; idx += 256) {
    int r = idx / XS1, c = idx % XS1, e = e0 + r;
    float v = 0.f;
    if (c == 0)        v = in_inf[e];
    else if (c < 65)   v = pld_re[e * ANT + (c - 1)];
    else if (c < 129)  v = pld_im[e * ANT + (c - 65)];
    else if (c < 193) {
      int j = c - 129;
      v = mlp_ue[e * D2 + j] + 128.f * b2b[j];
    }
    lds4[r * XS1 + c] = (_Float16)v;
  }
  const float bias0 = b1a[wv * 32 + (lane & 15)];
  const float bias1 = b1a[wv * 32 + 16 + (lane & 15)];
  __syncthreads();

  const f8 zero8 = {0.f, 0.f, 0.f, 0.f, 0.f, 0.f, 0.f, 0.f};
  f8 acc0 = zero8, acc1 = zero8;
#pragma unroll
  for (int kt = 0; kt < 7; ++kt) {
    h16 aF = frag_load(frag_ptr(lds4, 0, XS1, kt * 32, lane));
    h16 b0 = frag_load(frag_ptr(w1aT, wv * 32,      XS1, kt * 32, lane));
    h16 b1 = frag_load(frag_ptr(w1aT, wv * 32 + 16, XS1, kt * 32, lane));
    acc0 = WMMA_F16(aF, b0, acc0);
    acc1 = WMMA_F16(aF, b1, acc1);
  }
  const int hoff = K4_HS + wv * 16 * HSS;
  const int r0 = (lane < 16) ? 0 : 8, nl = lane & 15;
#pragma unroll
  for (int i = 0; i < 8; ++i) {
    float v0 = acc0[i] + bias0; v0 = v0 > 0.f ? v0 : 0.f;
    float v1 = acc1[i] + bias1; v1 = v1 > 0.f ? v1 : 0.f;
    lds4[hoff + (r0 + i) * HSS + nl]      = (_Float16)v0;
    lds4[hoff + (r0 + i) * HSS + 16 + nl] = (_Float16)v1;
  }
  h16 a2 = frag_load(frag_ptr(lds4 + hoff, 0, HSS, 0, lane));
#pragma unroll
  for (int nt = 0; nt < 8; ++nt) {
    h16 bF = frag_load(frag_ptr(w1bT, nt * 16, XS1, wv * 32, lane));
    f8 c = WMMA_F16(a2, bF, zero8);
#pragma unroll
    for (int i = 0; i < 8; ++i)
      atomicAdd(&outred[(r0 + i) * 128 + nt * 16 + nl], c[i]);
  }
  __syncthreads();
  for (int i = tid; i < 16 * 128; i += 256) {
    int m = i >> 7, d = i & 127, e = e0 + m;
    float v = outred[i] + b1b[d];
    if (d < 64) ue_re[e * ANT + d] = v;
    else        ue_im[e * ANT + (d - 64)] = v;
  }
}

// ---------------------------------------------------------------------------
// Kernel 5a: per-AP L1 normalize -> d_out (pv_new), plus |ss2|^2 precompute
// and zeroing of the interference output slots.
// ---------------------------------------------------------------------------
__global__ __launch_bounds__(128) void k5a(const float* __restrict__ ue_re,
                                           const float* __restrict__ ue_im,
                                           float* __restrict__ d_out,
                                           float* __restrict__ s2mag) {
  __shared__ float red[128];
  int a = blockIdx.x, tid = threadIdx.x;
  float s = 0.f;
  for (int j = tid; j < U_UE * ANT; j += 128) {
    float r = ue_re[a * 512 + j], i = ue_im[a * 512 + j];
    s += sqrtf(r * r + i * i);
  }
  red[tid] = s;
  __syncthreads();
  for (int off = 64; off > 0; off >>= 1) {
    if (tid < off) red[tid] += red[tid + off];
    __syncthreads();
  }
  float inv = 1.f / red[0];
  for (int j = tid; j < U_UE * ANT; j += 128) {
    d_out[a * 512 + j]              = ue_re[a * 512 + j] * inv;   // pv_new.re
    d_out[A_AP * 512 + a * 512 + j] = ue_im[a * 512 + j] * inv;   // pv_new.im
  }
  if (tid < ANT) {
    float sr = 0.f, si = 0.f;
    for (int u = 0; u < U_UE; ++u) {
      sr += ue_re[a * 512 + u * ANT + tid];
      si += ue_im[a * 512 + u * ANT + tid];
    }
    s2mag[a * ANT + tid] = (sr * sr + si * si) * inv * inv;
  }
  if (tid < U_UE) d_out[2 * A_AP * 512 + a * U_UE + tid] = 0.f;
}

// ---------------------------------------------------------------------------
// Kernel 5b: ue_all_interference[u] = sum_a sum_t |ss2[a][t]|^2 * |pl_int[e][t]|^2
// One wave (32 lanes, wave32) per edge; float2 loads (L2 hits from k2's pass);
// shfl reduce; atomicAdd.
// ---------------------------------------------------------------------------
__global__ __launch_bounds__(256) void k5b(const float* __restrict__ pl_re,
                                           const float* __restrict__ pl_im,
                                           const float* __restrict__ s2mag,
                                           float* __restrict__ out_int) {
  int tid = threadIdx.x, lane = tid & 31, wv = tid >> 5;
  int e = blockIdx.x * 8 + wv;   // 16384 blocks x 8 waves = 131072 edges
  int a = e >> 10, u = e & 1023;
  float2 r = ((const float2*)(pl_re + (size_t)e * ANT))[lane];
  float2 i = ((const float2*)(pl_im + (size_t)e * ANT))[lane];
  float s0 = s2mag[a * ANT + lane * 2];
  float s1 = s2mag[a * ANT + lane * 2 + 1];
  float v = s0 * (r.x * r.x + i.x * i.x) + s1 * (r.y * r.y + i.y * i.y);
  for (int off = 16; off > 0; off >>= 1) v += __shfl_down(v, off);
  if (lane == 0) atomicAdd(&out_int[u], v);
}

// ---------------------------------------------------------------------------
// Host-side launcher
// ---------------------------------------------------------------------------
extern "C" void kernel_launch(void* const* d_in, const int* in_sizes, int n_in,
                              void* d_out, int out_size, void* d_ws, size_t ws_size,
                              hipStream_t stream) {
  const float* pl_int_re = (const float*)d_in[0];
  const float* pl_int_im = (const float*)d_in[1];
  const float* pl_dl_re  = (const float*)d_in[2];
  const float* pl_dl_im  = (const float*)d_in[3];
  const float* pv_re     = (const float*)d_in[4];
  const float* pv_im     = (const float*)d_in[5];
  const float* W2a       = (const float*)d_in[6];
  const float* b2a       = (const float*)d_in[7];
  const float* W2b       = (const float*)d_in[8];
  const float* b2b       = (const float*)d_in[9];
  const float* W1a       = (const float*)d_in[10];
  const float* b1a       = (const float*)d_in[11];
  const float* W1b       = (const float*)d_in[12];
  const float* b1b       = (const float*)d_in[13];
  float* out = (float*)d_out;

  // Workspace carve-up (all offsets multiples of 256 bytes).
  char* ws = (char*)d_ws;
  _Float16* ssh    = (_Float16*)(ws + 0);        //  32768 B
  _Float16* w2aT   = (_Float16*)(ws + 32768);    // 139264 B  [256][272]
  _Float16* w2bT   = (_Float16*)(ws + 172032);   //  34816 B  [64][272]
  _Float16* w1aT   = (_Float16*)(ws + 206848);   // 122880 B  [256][240]
  _Float16* w1bT   = (_Float16*)(ws + 329728);   //  61440 B  [128][240]
  float*    mlp_ue = (float*)   (ws + 399360);   // 262144 B  [1024][64]
  float*    in_inf = (float*)   (ws + 661504);   //   4096 B
  float*    ue_re  = (float*)   (ws + 665600);   // 262144 B
  float*    ue_im  = (float*)   (ws + 927744);   // 262144 B
  float*    s2mag  = (float*)   (ws + 1189888);  //  32768 B

  // 1) AP sums + weight transposes / zeroing
  k1_ss<<<dim3(A_AP), dim3(64), 0, stream>>>(pv_re, pv_im, ssh);
  {
    int total = H * K2IN + D2 * H + H * K1PAD + (2 * ANT) * H + NUE * D2;
    k1_wt<<<dim3((total + 255) / 256), dim3(256), 0, stream>>>(
        W2a, W2b, W1a, W1b, w2aT, w2bT, w1aT, w1bT, mlp_ue);
  }

  // 2) dominant edge MLP + segment sum (WMMA, double-buffered pipeline)
  {
    size_t smem = (size_t)K2_OUT * 2 + (size_t)(16 * 64) * 4;   // ~31.7 KB
    k2_mlp2<<<dim3(256), dim3(256), smem, stream>>>(
        pl_int_re, pl_int_im, ssh, w2aT, w2bT, b2a, mlp_ue);
  }

  // 3) in-cell interference feature
  k3_ininf<<<dim3(32), dim3(256), 0, stream>>>(pv_re, pv_im, pl_dl_re, pl_dl_im, in_inf);

  // 4) per-UE MLP (WMMA)
  {
    size_t smem = (size_t)K4_OUT * 2 + (size_t)(16 * 128) * 4;  // ~26 KB
    k4_mlp1<<<dim3(NUE / 16), dim3(256), smem, stream>>>(
        pl_dl_re, pl_dl_im, in_inf, mlp_ue, b2b, b1a, b1b, w1aT, w1bT, ue_re, ue_im);
  }

  // 5) normalize + final interference
  k5a<<<dim3(A_AP), dim3(128), 0, stream>>>(ue_re, ue_im, out, s2mag);
  k5b<<<dim3(E_INT / 8), dim3(256), 0, stream>>>(
      pl_int_re, pl_int_im, s2mag, out + 2 * A_AP * 512);

  (void)in_sizes; (void)n_in; (void)out_size; (void)ws_size;
}
